// MultiHeadAttention_62577673503892
// MI455X (gfx1250) — compile-verified, matrix operand order checked by reading, not run on hardware
//
#include <hip/hip_runtime.h>
#include <hip/hip_bf16.h>

typedef __bf16 bf16;
typedef __attribute__((ext_vector_type(8)))  float  v8f;
typedef __attribute__((ext_vector_type(8)))  __bf16 v8bf;
typedef __attribute__((ext_vector_type(16))) __bf16 v16bf;

#define BATCH 2
#define SEQ   2048
#define DMODEL 1024
#define NHEAD 16
#define HDIM  64
#define ME    (1024*1024)   // one 1024x1024 matrix, element count

__device__ __forceinline__ v8f vzero8() {
  v8f z;
#pragma unroll
  for (int i = 0; i < 8; ++i) z[i] = 0.0f;
  return z;
}

// A/B fragment (16-bit, 16x32 per ISA 7.12.2): lane half selects 8+8 element runs.
__device__ __forceinline__ v16bf load_bf_frag(const bf16* base, int half) {
  v8bf lo = *(const v8bf*)(base + half * 8);
  v8bf hi = *(const v8bf*)(base + 16 + half * 8);
  v16bf r;
#pragma unroll
  for (int i = 0; i < 8; ++i) { r[i] = lo[i]; r[i + 8] = hi[i]; }
  return r;
}

__device__ __forceinline__ v16bf load_f32_frag(const float* base, int half) {
  const float* p0 = base + half * 8;
  const float* p1 = base + 16 + half * 8;
  v16bf r;
#pragma unroll
  for (int i = 0; i < 8; ++i) { r[i] = (bf16)p0[i]; r[i + 8] = (bf16)p1[i]; }
  return r;
}

__device__ __forceinline__ v8f wmma_bf16(v16bf a, v16bf b, v8f c) {
  return __builtin_amdgcn_wmma_f32_16x16x32_bf16(false, a, false, b, (short)0, c,
                                                 false, false);
}

// ---------------------------------------------------------------------------
// Kernel 0: f32 [K][N] -> bf16 [N][K] transpose for Wq,Wk,Wv,Wfc
// ---------------------------------------------------------------------------
__global__ __launch_bounds__(256) void transpose_cvt(
    const float* __restrict__ W0, const float* __restrict__ W1,
    const float* __restrict__ W2, const float* __restrict__ W3,
    bf16* __restrict__ out) {
  __shared__ float tile[32][33];
  const float* W = (blockIdx.z == 0) ? W0 : (blockIdx.z == 1) ? W1
                   : (blockIdx.z == 2) ? W2 : W3;
  bf16* T = out + (size_t)blockIdx.z * ME;
  int x  = blockIdx.x * 32 + threadIdx.x;   // source col n
  int y0 = blockIdx.y * 32;                 // source row k base
#pragma unroll
  for (int j = threadIdx.y; j < 32; j += 8)
    tile[j][threadIdx.x] = W[(size_t)(y0 + j) * DMODEL + x];
  __syncthreads();
#pragma unroll
  for (int j = threadIdx.y; j < 32; j += 8) {
    int n = blockIdx.x * 32 + j;
    int k = y0 + threadIdx.x;
    T[(size_t)n * DMODEL + k] = (bf16)tile[threadIdx.x][j];
  }
}

// ---------------------------------------------------------------------------
// Kernel 1: Q/K/V projections.  X[4096x1024]f32 * W -> bf16.
//   Q,K stored head-split [B,H,L,64] (Q pre-scaled by 1/sqrt(64)).
//   V stored TRANSPOSED [B,H,64,L]: C-fragment rows are consecutive l, so the
//   V epilogue is 4 aligned v8bf (b128) vector stores per wave.
// One 16x64 output tile per wave; 128 WMMAs per wave.  Prefetches are
// unconditional (speculative prefetch past a region is dropped, ISA 10.5).
// ---------------------------------------------------------------------------
__global__ __launch_bounds__(256) void gemm_qkv(
    const float* __restrict__ qi, const float* __restrict__ ki,
    const float* __restrict__ vi, const bf16* __restrict__ WT,
    bf16* __restrict__ qh, bf16* __restrict__ kh, bf16* __restrict__ vhT) {
  const int which = blockIdx.z;
  const float* X = (which == 0) ? qi : (which == 1) ? ki : vi;
  const bf16* Wt = WT + (size_t)which * ME;
  bf16* O        = (which == 0) ? qh : (which == 1) ? kh : vhT;

  const int wave = threadIdx.x >> 5, lane = threadIdx.x & 31;
  const int half = lane >> 4, n16 = lane & 15;
  const int mtile = blockIdx.x * 8 + wave;     // 16-row tile, 0..255
  const int nbase = blockIdx.y * 64;
  const float* xrow = X + (size_t)(mtile * 16 + n16) * DMODEL;

  v8f acc[4];
#pragma unroll
  for (int t = 0; t < 4; ++t) acc[t] = vzero8();

  for (int kb = 0; kb < DMODEL; kb += 32) {
    __builtin_prefetch(xrow + kb + 64, 0, 1);   // global_prefetch_b8
    v16bf a = load_f32_frag(xrow + kb, half);
#pragma unroll
    for (int t = 0; t < 4; ++t) {
      const bf16* wr = Wt + (size_t)(nbase + t * 16 + n16) * DMODEL + kb;
      v16bf b = load_bf_frag(wr, half);
      acc[t] = wmma_bf16(a, b, acc[t]);
    }
  }

  // Epilogue: uniform `which` branch hoisted out of the store loops.
  // C layout: element i -> row M = i + 8*half, col N = lane&15.
  const int gmb   = mtile * 16;
  const int bidx  = gmb >> 11;            // batch (tiles never straddle it)
  const int lbase = (gmb & 2047) + 8 * half;
  if (which == 2) {                       // V: [B,H,D,L]; contiguous along l
#pragma unroll
    for (int t = 0; t < 4; ++t) {
      int col = nbase + t * 16 + n16;
      int h = col >> 6, dd = col & 63;
      bf16* op = O + (((size_t)bidx * NHEAD + h) * HDIM + dd) * SEQ + lbase;
      v8bf pk;
#pragma unroll
      for (int i = 0; i < 8; ++i) pk[i] = (bf16)acc[t][i];
      *(v8bf*)op = pk;                    // one aligned b128 store
    }
  } else {                                // Q,K: [B,H,L,D]; stride HDIM over l
    const float scale = (which == 0) ? 0.125f : 1.0f;
#pragma unroll
    for (int t = 0; t < 4; ++t) {
      int col = nbase + t * 16 + n16;
      int h = col >> 6, dd = col & 63;
      bf16* op = O + (((size_t)bidx * NHEAD + h) * SEQ + lbase) * HDIM + dd;
#pragma unroll
      for (int i = 0; i < 8; ++i)
        op[(size_t)i * HDIM] = (bf16)(acc[t][i] * scale);
    }
  }
}

// ---------------------------------------------------------------------------
// Kernel 2: fused causal attention per (b,h,16-row block):
// S = Q*K^T (WMMA, causally skipped tiles) -> LDS; softmax (wave32 shuffles);
// write attn f32 once to HBM + P bf16 to LDS; ctx = P*V^T (WMMA, K split 2x).
// ---------------------------------------------------------------------------
#define SSTR 2064   // padded score row stride (f32 / bf16 elements)
#define CSTR 68     // padded ctx staging stride

__global__ __launch_bounds__(256) void attn_kernel(
    const bf16* __restrict__ qh, const bf16* __restrict__ kh,
    const bf16* __restrict__ vhT, float* __restrict__ attn_out,
    bf16* __restrict__ ctx) {
  extern __shared__ char smem[];
  float* S  = (float*)smem;                                    // [16][SSTR]
  bf16*  P  = (bf16*)(smem + 16 * SSTR * 4);                   // [16][SSTR]
  float* CT = (float*)(smem + 16 * SSTR * 4 + 16 * SSTR * 2);  // [16][CSTR]

  const int bh = blockIdx.y;                  // b*H + h
  const int rb = blockIdx.x * 16;             // query row base
  const int wave = threadIdx.x >> 5, lane = threadIdx.x & 31;
  const int half = lane >> 4, n16 = lane & 15;

  const bf16* Q  = qh  + ((size_t)bh * SEQ + rb) * HDIM;
  const bf16* K  = kh  + (size_t)bh * SEQ * HDIM;
  const bf16* Vt = vhT + (size_t)bh * HDIM * SEQ;   // [64][2048]

  // Q fragments for K-dim = 64 (two 32-chunks); Q already scaled by 1/8.
  v16bf aq[2];
#pragma unroll
  for (int c = 0; c < 2; ++c)
    aq[c] = load_bf_frag(Q + (size_t)n16 * HDIM + c * 32, half);

  // ---- scores: each wave owns 256 key columns (16 tiles) ----
  for (int t = 0; t < 16; ++t) {
    int cb = wave * 256 + t * 16;
    if (cb > rb + 15) continue;               // fully masked tile (wave-uniform)
    __builtin_prefetch(K + (size_t)(cb + 16 + n16) * HDIM, 0, 1);
    v8f c = vzero8();
#pragma unroll
    for (int kc = 0; kc < 2; ++kc) {
      v16bf b = load_bf_frag(K + (size_t)(cb + n16) * HDIM + kc * 32, half);
      c = wmma_bf16(aq[kc], b, c);
    }
#pragma unroll
    for (int i = 0; i < 8; ++i)
      S[(i + 8 * half) * SSTR + cb + n16] = c[i];
  }
  __syncthreads();

  // ---- softmax: 2 rows per wave, causal mask applied here ----
  const size_t attn_base = ((size_t)bh * SEQ + rb) * SEQ;
#pragma unroll
  for (int rr = 0; rr < 2; ++rr) {
    int r = wave * 2 + rr;
    int ar = rb + r;                          // valid keys: 0..ar
    float m = -1e30f;
    for (int c0 = lane; c0 < SEQ; c0 += 32) {
      float sv = -1e9f;
      if (c0 <= ar) sv = S[r * SSTR + c0];
      m = fmaxf(m, sv);
    }
#pragma unroll
    for (int o = 16; o >= 1; o >>= 1) m = fmaxf(m, __shfl_xor(m, o, 32));
    float sum = 0.0f;
    for (int c0 = lane; c0 < SEQ; c0 += 32) {
      float p = 0.0f;
      if (c0 <= ar) p = __expf(S[r * SSTR + c0] - m);
      S[r * SSTR + c0] = p;
      sum += p;
    }
#pragma unroll
    for (int o = 16; o >= 1; o >>= 1) sum += __shfl_xor(sum, o, 32);
    float rinv = 1.0f / sum;                  // sum >= exp(0) = 1 (diagonal)
    for (int c0 = lane; c0 < SEQ; c0 += 32) {
      float p = S[r * SSTR + c0] * rinv;
      attn_out[attn_base + (size_t)r * SEQ + c0] = p;
      P[r * SSTR + c0] = (bf16)p;
    }
  }
  __syncthreads();

  // ---- ctx = P * V^T : waves (g, ntile) = (w>>2, w&3); causal K bound ----
  const int ntile = wave & 3;
  const int g = wave >> 2;
  const int kend = rb + 16;                   // keys beyond block are zero
  int k0 = g * 1024;
  int k1 = (g + 1) * 1024; if (k1 > kend) k1 = kend;
  const int d = ntile * 16 + n16;
  const bf16* vrow = Vt + (size_t)d * SEQ;    // contiguous over keys
  v8f c = vzero8();
  for (int kb = k0; kb < k1; kb += 32) {
    __builtin_prefetch(vrow + kb + 32, 0, 1);
    v16bf a = load_bf_frag(P + (size_t)n16 * SSTR + kb, half);
    v16bf b = load_bf_frag(vrow + kb, half);
    c = wmma_bf16(a, b, c);
  }
  if (g == 0) {
#pragma unroll
    for (int i = 0; i < 8; ++i) CT[(i + 8 * half) * CSTR + d] = c[i];
  }
  __syncthreads();
  if (g == 1) {
    const int b_ = bh >> 4, h = bh & 15;
    bf16* cp = ctx + ((size_t)b_ * SEQ + rb + 8 * half) * DMODEL + h * HDIM + d;
#pragma unroll
    for (int i = 0; i < 8; ++i)
      cp[(size_t)i * DMODEL] = (bf16)(c[i] + CT[(i + 8 * half) * CSTR + d]);
  }
}

// ---------------------------------------------------------------------------
// Kernel 3: out = LN(ctx*Wfc + residual).  One 16-row block per workgroup;
// each wave owns 8 N-tiles (128 cols), K-outer loop reuses the A fragment.
// ---------------------------------------------------------------------------
#define YS 1032

__global__ __launch_bounds__(256) void proj_ln(
    const bf16* __restrict__ ctx, const bf16* __restrict__ WfcT,
    const float* __restrict__ res, const float* __restrict__ gamma,
    const float* __restrict__ beta, float* __restrict__ out) {
  extern __shared__ char smem[];
  float* Y = (float*)smem;                    // [16][YS]
  const int mb = blockIdx.x * 16;
  const int wave = threadIdx.x >> 5, lane = threadIdx.x & 31;
  const int half = lane >> 4, n16 = lane & 15;

  const bf16* arow = ctx + (size_t)(mb + n16) * DMODEL;
  v8f acc[8];
#pragma unroll
  for (int t = 0; t < 8; ++t) acc[t] = vzero8();

  for (int kb = 0; kb < DMODEL; kb += 32) {
    __builtin_prefetch(arow + kb + 32, 0, 1);
    v16bf a = load_bf_frag(arow + kb, half);
#pragma unroll
    for (int t = 0; t < 8; ++t) {
      const bf16* wr = WfcT + (size_t)(wave * 128 + t * 16 + n16) * DMODEL + kb;
      v16bf b = load_bf_frag(wr, half);
      acc[t] = wmma_bf16(a, b, acc[t]);
    }
  }
#pragma unroll
  for (int t = 0; t < 8; ++t)
#pragma unroll
    for (int i = 0; i < 8; ++i)
      Y[(i + 8 * half) * YS + wave * 128 + t * 16 + n16] = acc[t][i];
  __syncthreads();

  // residual + LayerNorm, 2 rows per wave
#pragma unroll
  for (int rr = 0; rr < 2; ++rr) {
    int r = wave * 2 + rr;
    int m = mb + r;
    const float* qrow = res + (size_t)m * DMODEL;
    float s = 0.0f, s2 = 0.0f;
    for (int c0 = lane; c0 < DMODEL; c0 += 32) {
      float y = Y[r * YS + c0] + qrow[c0];
      Y[r * YS + c0] = y;
      s += y; s2 += y * y;
    }
#pragma unroll
    for (int o = 16; o >= 1; o >>= 1) { s += __shfl_xor(s, o, 32); s2 += __shfl_xor(s2, o, 32); }
    float mu  = s * (1.0f / DMODEL);
    float var = s2 * (1.0f / DMODEL) - mu * mu;
    float rs  = rsqrtf(var + 1e-6f);
    for (int c0 = lane; c0 < DMODEL; c0 += 32) {
      float y = Y[r * YS + c0];
      out[(size_t)m * DMODEL + c0] = (y - mu) * rs * gamma[c0] + beta[c0];
    }
  }
}

// ---------------------------------------------------------------------------
extern "C" void kernel_launch(void* const* d_in, const int* in_sizes, int n_in,
                              void* d_out, int out_size, void* d_ws, size_t ws_size,
                              hipStream_t stream) {
  const float* q     = (const float*)d_in[0];
  const float* k     = (const float*)d_in[1];
  const float* v     = (const float*)d_in[2];
  const float* Wq    = (const float*)d_in[3];
  const float* Wk    = (const float*)d_in[4];
  const float* Wv    = (const float*)d_in[5];
  const float* Wfc   = (const float*)d_in[6];
  const float* gamma = (const float*)d_in[7];
  const float* beta  = (const float*)d_in[8];
  // d_in[9] = mask: causal tril, applied analytically in attn_kernel.

  float* out  = (float*)d_out;
  float* attn = out + (size_t)BATCH * SEQ * DMODEL;   // outputs concatenated

  bf16* ws  = (bf16*)d_ws;
  bf16* WT  = ws;                  // 4x [1024][1024] bf16 transposed weights
  bf16* qh  = ws + (size_t)4  * ME;
  bf16* kh  = ws + (size_t)8  * ME;
  bf16* vhT = ws + (size_t)12 * ME;
  bf16* ctx = ws + (size_t)16 * ME;

  dim3 bt(32, 8), gt(32, 32, 4);
  transpose_cvt<<<gt, bt, 0, stream>>>(Wq, Wk, Wv, Wfc, WT);

  dim3 gg(32, 16, 3);
  gemm_qkv<<<gg, 256, 0, stream>>>(q, k, v, WT, qh, kh, vhT);

  dim3 ga(SEQ / 16, BATCH * NHEAD);
  size_t lds_a = 16 * SSTR * 4 + 16 * SSTR * 2 + 16 * CSTR * 4;  // ~202 KB
  attn_kernel<<<ga, 256, lds_a, stream>>>(qh, kh, vhT, attn, ctx);

  size_t lds_p = 16 * YS * 4;                                    // ~66 KB
  proj_ln<<<(BATCH * SEQ) / 16, 256, lds_p, stream>>>(ctx, WT + (size_t)3 * ME,
                                                      q, gamma, beta, out);
}
